// MambaBlock_13030930776224
// MI455X (gfx1250) — matrix operand order repaired and checked by reading, not compiled, not run on hardware
//
#include <hip/hip_runtime.h>
#include <math.h>

#define DIMX 1024
#define DI   2048
#define NST  16
#define DTRK 64
#define DCONV 4
#define BB   2
#define LL   1024

#define TM_BLK 64
#define TN_BLK 128
#define TK_BLK 32

typedef __attribute__((ext_vector_type(16))) _Float16 v16h;
typedef __attribute__((ext_vector_type(8)))  float    v8f;

__device__ __forceinline__ float sigmoidf_(float x) { return 1.0f / (1.0f + __expf(-x)); }

// CDNA5 async global->LDS copy, 16 bytes per lane, tracked by ASYNCcnt.
__device__ __forceinline__ void async_copy_b128(unsigned lds_off, const void* gaddr) {
    asm volatile("global_load_async_to_lds_b128 %0, %1, off"
                 :: "v"(lds_off), "v"(gaddr)
                 : "memory");
}
__device__ __forceinline__ void wait_asynccnt0() {
    asm volatile("s_wait_asynccnt 0x0" ::: "memory");
}

// ---------------------------------------------------------------------------
// One-time weight prep: W (KxN f32, row-major) -> Wt (NxK f16, row-major),
// i.e. f16 transpose.  Makes the GEMM B fragment a contiguous 32B LDS read.
// ---------------------------------------------------------------------------
__global__ __launch_bounds__(256) void transpose_f16_kernel(
    const float* __restrict__ W, _Float16* __restrict__ Wt, int K, int N)
{
    int idx = blockIdx.x * 256 + threadIdx.x;
    if (idx >= K * N) return;
    const int k = idx / N;
    const int n = idx % N;
    Wt[(size_t)n * K + k] = (_Float16)W[idx];
}

// ---------------------------------------------------------------------------
// WMMA GEMM: D(MxN) = A(MxK) * B(KxN);  A fp32 row-major, B pre-transposed
// f16 (NxK).  Block tile 64x128, K step 32; A (f32) and Bt (f16) tiles staged
// to LDS with async b128 copies.  8 waves; wave (wr,wc) owns a 16x64 strip:
// 1 A fragment reused across 4 back-to-back v_wmma_f32_16x16x32_f16.
// epilogue==1: D = softplus(acc + bias[col])
// ---------------------------------------------------------------------------
__global__ __launch_bounds__(256) void wmma_gemm_kernel(
    const float* __restrict__ A, const _Float16* __restrict__ Bt,
    float* __restrict__ D, int M, int N, int K,
    int lda, int ldd,
    const float* __restrict__ bias, int epilogue)
{
    __shared__ __align__(16) float    sA [TM_BLK * TK_BLK];   // 64x32 f32 = 8KB
    __shared__ __align__(16) _Float16 sBt[TN_BLK * TK_BLK];   // 128x32 f16 = 8KB

    const int tid  = threadIdx.x;
    const int lane = tid & 31;
    const int wave = tid >> 5;
    const int nblk = (N + TN_BLK - 1) / TN_BLK;
    const int bm   = blockIdx.x / nblk;
    const int bn   = blockIdx.x % nblk;
    const int row0 = bm * TM_BLK;
    const int col0 = bn * TN_BLK;

    const int wr = wave & 3;     // 16-row strip
    const int wc = wave >> 2;    // 64-col strip
    const int hi = lane >> 4;
    const int lo = lane & 15;

    v8f acc[4] = {};

    for (int k0 = 0; k0 < K; k0 += TK_BLK) {
        __syncthreads();   // previous fragments consumed

        // stage A tile: 64x32 f32 -> 512 16B chunks, 2 per thread
#pragma unroll
        for (int c = 0; c < 2; ++c) {
            const int chunk = tid * 2 + c;
            const int r  = chunk >> 3;          // 0..63
            const int cc = (chunk & 7) * 4;     // 0,4,..,28
            async_copy_b128((unsigned)(uintptr_t)&sA[r * TK_BLK + cc],
                            A + (size_t)(row0 + r) * lda + k0 + cc);
        }
        // stage Bt tile: 128 cols x 32 K f16 -> 512 16B chunks, 2 per thread
#pragma unroll
        for (int c = 0; c < 2; ++c) {
            const int chunk = tid * 2 + c;
            const int col = chunk >> 2;         // 0..127
            const int kc  = (chunk & 3) * 8;    // 0,8,16,24 (halves)
            if (col0 + col < N)
                async_copy_b128((unsigned)(uintptr_t)&sBt[col * TK_BLK + kc],
                                Bt + (size_t)(col0 + col) * K + k0 + kc);
        }
        wait_asynccnt0();
        __syncthreads();

        // A fragment (16x32 f16): lane holds K = hi*16..+15
        v16h a;
        {
            const float* p0 = &sA[(wr * 16 + lo) * TK_BLK + hi * 16];
            const float* p1 = p0 + 8;
#pragma unroll
            for (int j = 0; j < 8; ++j) a[j]     = (_Float16)p0[j];
#pragma unroll
            for (int j = 0; j < 8; ++j) a[8 + j] = (_Float16)p1[j];
        }
        // all 4 B fragments first (contiguous 32B LDS reads), then 4 WMMAs
        v16h bfrag[4];
#pragma unroll
        for (int nt = 0; nt < 4; ++nt) {
            const int col = wc * 64 + nt * 16 + lo;
            bfrag[nt] = *(const v16h*)&sBt[col * TK_BLK + hi * 16];
        }
#pragma unroll
        for (int nt = 0; nt < 4; ++nt) {
            acc[nt] = __builtin_amdgcn_wmma_f32_16x16x32_f16(
                false, a, false, bfrag[nt], (short)0, acc[nt], false, false);
        }
    }

    // store: acc VGPR i -> row = i + 8*hi, lane lo -> col
    if (epilogue == 1) {
#pragma unroll
        for (int nt = 0; nt < 4; ++nt) {
            const int col = col0 + wc * 64 + nt * 16 + lo;
            if (col < N) {
                const float bv = bias[col];
#pragma unroll
                for (int i = 0; i < 8; ++i) {
                    const int row = row0 + wr * 16 + i + 8 * hi;
                    float v = acc[nt][i] + bv;
                    v = (v > 20.0f) ? v : __logf(1.0f + __expf(v));   // softplus
                    D[(size_t)row * ldd + col] = v;
                }
            }
        }
    } else {
#pragma unroll
        for (int nt = 0; nt < 4; ++nt) {
            const int col = col0 + wc * 64 + nt * 16 + lo;
            if (col < N) {
#pragma unroll
                for (int i = 0; i < 8; ++i) {
                    const int row = row0 + wr * 16 + i + 8 * hi;
                    D[(size_t)row * ldd + col] = acc[nt][i];
                }
            }
        }
    }
}

// ---------------------------------------------------------------------------
// Depthwise causal conv (k=4, left pad 3) + bias + SiLU.
// ---------------------------------------------------------------------------
__global__ __launch_bounds__(256) void conv_silu_kernel(
    const float* __restrict__ xz, const float* __restrict__ w,
    const float* __restrict__ bias, float* __restrict__ out)
{
    int idx = blockIdx.x * blockDim.x + threadIdx.x;   // (b, l, ch), ch fastest
    if (idx >= BB * LL * DI) return;
    const int ch = idx % DI;
    const int l  = (idx / DI) % LL;
    const int b  = idx / (DI * LL);

    float acc = bias[ch];
#pragma unroll
    for (int t = 0; t < DCONV; ++t) {
        const int ls = l - (DCONV - 1) + t;
        if (ls >= 0)
            acc += w[ch * DCONV + t] * xz[(size_t)(b * LL + ls) * (2 * DI) + ch];
    }
    out[(size_t)idx] = acc * sigmoidf_(acc);
}

// ---------------------------------------------------------------------------
// Selective scan: 16-lane half-wave per (b, channel); lane = state n.
// ---------------------------------------------------------------------------
__global__ __launch_bounds__(256) void selective_scan_kernel(
    const float* __restrict__ u,      // (B*L, DI)
    const float* __restrict__ delta,  // (B*L, DI)
    const float* __restrict__ xdbl,   // (B*L, 96)
    const float* __restrict__ A_log,  // (DI, 16)
    const float* __restrict__ Dvec,   // (DI)
    const float* __restrict__ xz,     // (B*L, 2*DI)
    float* __restrict__ yo)           // (B*L, DI)
{
    const int half = (blockIdx.x * blockDim.x + threadIdx.x) >> 4;
    const int n    = threadIdx.x & 15;
    if (half >= BB * DI) return;
    const int b  = half / DI;
    const int ch = half % DI;

    const float An  = -__expf(A_log[ch * NST + n]);
    const float Dch = Dvec[ch];
    float h = 0.0f;

    for (int l = 0; l < LL; ++l) {
        const size_t row = (size_t)b * LL + l;
        const float dlt = delta[row * DI + ch];
        const float uu  = u[row * DI + ch];
        const float Bn  = xdbl[row * 96 + DTRK + n];
        const float Cn  = xdbl[row * 96 + DTRK + NST + n];

        const float dA = __expf(dlt * An);
        h = dA * h + dlt * Bn * uu;

        float p = h * Cn;
        p += __shfl_xor(p, 8, 16);
        p += __shfl_xor(p, 4, 16);
        p += __shfl_xor(p, 2, 16);
        p += __shfl_xor(p, 1, 16);

        if (n == 0) {
            const float r = xz[row * (2 * DI) + DI + ch];
            yo[row * DI + ch] = (p + uu * Dch) * (r * sigmoidf_(r));
        }
    }
}

// ---------------------------------------------------------------------------
extern "C" void kernel_launch(void* const* d_in, const int* in_sizes, int n_in,
                              void* d_out, int out_size, void* d_ws, size_t ws_size,
                              hipStream_t stream)
{
    const float* x       = (const float*)d_in[0];
    const float* W_in    = (const float*)d_in[1];
    const float* conv_w  = (const float*)d_in[2];
    const float* conv_b  = (const float*)d_in[3];
    const float* W_xproj = (const float*)d_in[4];
    const float* W_dt    = (const float*)d_in[5];
    const float* b_dt    = (const float*)d_in[6];
    const float* A_log   = (const float*)d_in[7];
    const float* Dvec    = (const float*)d_in[8];
    const float* W_out   = (const float*)d_in[9];
    float*       out     = (float*)d_out;

    // f32 workspace
    float* ws  = (float*)d_ws;
    size_t off = 0;
    float* xz    = ws + off; off += (size_t)BB * LL * 2 * DI;
    float* xconv = ws + off; off += (size_t)BB * LL * DI;
    float* xdbl  = ws + off; off += (size_t)BB * LL * 96;
    float* delta = ws + off; off += (size_t)BB * LL * DI;
    float* yo    = ws + off; off += (size_t)BB * LL * DI;
    // f16 transposed weights (region starts 16B-aligned)
    _Float16* h16 = (_Float16*)(ws + off);
    size_t hoff = 0;
    _Float16* W_in_t    = h16 + hoff; hoff += (size_t)(2 * DI) * DIMX;  // 4096x1024
    _Float16* W_xproj_t = h16 + hoff; hoff += (size_t)96 * DI;          // 96x2048
    _Float16* W_dt_t    = h16 + hoff; hoff += (size_t)DI * DTRK;        // 2048x64
    _Float16* W_out_t   = h16 + hoff; hoff += (size_t)DIMX * DI;        // 1024x2048

    const int M = BB * LL;   // 2048
    const dim3 blk(256);
    auto nblocks = [](int m, int n) {
        return (m / TM_BLK) * ((n + TN_BLK - 1) / TN_BLK);
    };
    auto tblocks = [](int k, int n) { return (k * n + 255) / 256; };

    // ---- one-time weight prep: f16 transpose of all GEMM B operands ----
    transpose_f16_kernel<<<tblocks(DIMX, 2 * DI), blk, 0, stream>>>(W_in,    W_in_t,    DIMX, 2 * DI);
    transpose_f16_kernel<<<tblocks(DI, 96),       blk, 0, stream>>>(W_xproj, W_xproj_t, DI,   96);
    transpose_f16_kernel<<<tblocks(DTRK, DI),     blk, 0, stream>>>(W_dt,    W_dt_t,    DTRK, DI);
    transpose_f16_kernel<<<tblocks(DI, DIMX),     blk, 0, stream>>>(W_out,   W_out_t,   DI,   DIMX);

    // G1: xz = x @ W_in                    (2048x1024)x(1024x4096)
    wmma_gemm_kernel<<<nblocks(M, 2 * DI), blk, 0, stream>>>(
        x, W_in_t, xz, M, 2 * DI, DIMX, DIMX, 2 * DI, nullptr, 0);

    // conv + SiLU on x-half of xz -> xconv
    {
        int n = BB * LL * DI;
        conv_silu_kernel<<<(n + 255) / 256, blk, 0, stream>>>(xz, conv_w, conv_b, xconv);
    }

    // G2: xdbl = xconv @ W_xproj           (2048x2048)x(2048x96)
    wmma_gemm_kernel<<<nblocks(M, 96), blk, 0, stream>>>(
        xconv, W_xproj_t, xdbl, M, 96, DI, DI, 96, nullptr, 0);

    // G3: delta = softplus(xdbl[:, :64] @ W_dt + b_dt)   (2048x64)x(64x2048)
    wmma_gemm_kernel<<<nblocks(M, DI), blk, 0, stream>>>(
        xdbl, W_dt_t, delta, M, DI, DTRK, 96, DI, b_dt, 1);

    // selective scan (fused * silu(res))
    {
        int threads = BB * DI * NST;   // 65536
        selective_scan_kernel<<<threads / 256, blk, 0, stream>>>(
            xconv, delta, xdbl, A_log, Dvec, xz, yo);
    }

    // G4: out = yo @ W_out                 (2048x2048)x(2048x1024)
    wmma_gemm_kernel<<<nblocks(M, DIMX), blk, 0, stream>>>(
        yo, W_out_t, out, M, DIMX, DI, DI, DIMX, nullptr, 0);
}
